// ConstrainedEnhancementModel_9466107920637
// MI455X (gfx1250) — compile-verified, matrix operand order checked by reading, not run on hardware
//
#include <hip/hip_runtime.h>

typedef __attribute__((ext_vector_type(16))) _Float16 v16h;
typedef __attribute__((ext_vector_type(8)))  _Float16 h8;
typedef __attribute__((ext_vector_type(8)))  float    v8f;
typedef __attribute__((ext_vector_type(4)))  int      i4;

#define LOW_T   100
#define HIGH_T  2000
#define FEAT    6
#define UPF     20
#define BATCH   4096

// ---------------------------------------------------------------------------
// CDNA5 async global->LDS copy (ASYNCcnt-tracked), builtin if available,
// inline asm fallback otherwise.
// ---------------------------------------------------------------------------
#if defined(__has_builtin)
#if __has_builtin(__builtin_amdgcn_global_load_async_to_lds_b128)
#define HAVE_ASYNC_BUILTIN 1
#endif
#if __has_builtin(__builtin_amdgcn_s_wait_asynccnt)
#define HAVE_WAIT_BUILTIN 1
#endif
#endif

// ROCm clang signature: (int4 AS1* src, int4 AS3* dst, imm off, imm cpol)
typedef __attribute__((address_space(1))) i4* as1_i4p;
typedef __attribute__((address_space(3))) i4* as3_i4p;

__device__ __forceinline__ void async_copy16(const void* g, const void* l)
{
#ifdef HAVE_ASYNC_BUILTIN
    __builtin_amdgcn_global_load_async_to_lds_b128(
        (as1_i4p)(unsigned long long)g,
        (as3_i4p)(unsigned)(unsigned long long)l,
        0, 0);
#else
    // VDST = LDS byte offset (low 32 bits of flat LDS address), VADDR = 64-bit global
    asm volatile("global_load_async_to_lds_b128 %0, %1, off"
                 :: "v"((unsigned)(unsigned long long)l),
                    "v"((unsigned long long)g)
                 : "memory");
#endif
}

#ifdef HAVE_WAIT_BUILTIN
#define WAIT_ASYNC(n) __builtin_amdgcn_s_wait_asynccnt(n)
#else
#define WAIT_ASYNC(n) asm volatile("s_wait_asynccnt %0" :: "i"(n) : "memory")
#endif

// ---------------------------------------------------------------------------
// f32 -> f16 conversion with zero padding (row-major, same orientation).
// Used for the activation input X (600 -> 640 K-padding).
// ---------------------------------------------------------------------------
__global__ __launch_bounds__(256) void pad_cast_f16(
    const float* __restrict__ src, _Float16* __restrict__ dst,
    int Rs, int Cs, int Rd, int Cd)
{
    int i = blockIdx.x * 256 + threadIdx.x;
    int total = Rd * Cd;
    if (i >= total) return;
    int r = i / Cd;
    int c = i - r * Cd;
    float v = (r < Rs && c < Cs) ? src[(size_t)r * Cs + c] : 0.0f;
    dst[i] = (_Float16)v;
}

// ---------------------------------------------------------------------------
// Transposing f32 -> f16 converter: src (Ks x N) -> dst (N x Kd), Kd >= Ks
// zero-padded. LDS-tiled so both global sides are coalesced.
// Requires N % 32 == 0 and Kd % 32 == 0 (true for all layers).
// block (32,8), grid (N/32, Kd/32).
// ---------------------------------------------------------------------------
__global__ __launch_bounds__(256) void transpose_cast_f16(
    const float* __restrict__ src, _Float16* __restrict__ dst,
    int Ks, int N, int Kd)
{
    __shared__ float tile[32][33];
    const int tx = threadIdx.x;       // 0..31
    const int ty = threadIdx.y;       // 0..7
    const int n0 = blockIdx.x * 32;
    const int k0 = blockIdx.y * 32;
#pragma unroll
    for (int j = 0; j < 4; ++j) {
        int k = k0 + ty + j * 8;
        float v = (k < Ks) ? src[(size_t)k * N + (n0 + tx)] : 0.0f;
        tile[ty + j * 8][tx] = v;
    }
    __syncthreads();
#pragma unroll
    for (int j = 0; j < 4; ++j) {
        int n = n0 + ty + j * 8;
        dst[(size_t)n * Kd + (k0 + tx)] = (_Float16)tile[tx][ty + j * 8];
    }
}

// ---------------------------------------------------------------------------
// Fill columns [128,160) of the 4096x160 decoder-input matrix:
// cols 128..143 = emb[labels[b]], cols 144..159 = 0 (K padding).
// ---------------------------------------------------------------------------
__global__ __launch_bounds__(256) void feat_extra_kernel(
    const int* __restrict__ labels, const float* __restrict__ emb,
    _Float16* __restrict__ F)
{
    int i = blockIdx.x * 256 + threadIdx.x;   // BATCH * 32 elements
    if (i >= BATCH * 32) return;
    int b = i >> 5;
    int j = i & 31;
    float v = 0.0f;
    if (j < 16) v = emb[labels[b] * 16 + j];
    F[(size_t)b * 160 + 128 + j] = (_Float16)v;
}

// ---------------------------------------------------------------------------
// WMMA GEMM:  C(MxN) = act(A(MxK,f16) @ Wt^T + bias),  Wt is N x K (f16).
// A row stride == K, Wt row stride == K (all layers padded so this holds).
//   MODE 0: ReLU, store f16 into C (ldc stride)
//   MODE 1: no act, store f16 into C (ldc stride)
//   MODE 2: final layer -> fused interpolation/blend epilogue, f32 into Out
// Block: 256 threads (8 waves), tile 128(M) x 64(N), K stepped by 32,
// double-buffered LDS filled by GLOBAL_LOAD_ASYNC_TO_LDS_B128.
// ---------------------------------------------------------------------------
template <int MODE>
__global__ __launch_bounds__(256) void gemm_wmma(
    const _Float16* __restrict__ A, const _Float16* __restrict__ Wt,
    const float* __restrict__ bias, _Float16* __restrict__ C,
    const float* __restrict__ low, float* __restrict__ Out,
    int N, int K, int ldc)
{
    // per buffer: A = 128 rows x 64B (32 f16), B = 64 n-rows x 64B (32 f16)
    __shared__ __align__(16) unsigned char ldsA[2][128 * 64];
    __shared__ __align__(16) unsigned char ldsB[2][64 * 64];

    const int tid   = threadIdx.x;
    const int lane  = tid & 31;
    const int wave  = tid >> 5;
    const int g     = lane >> 4;     // K-half selector per WMMA layout
    const int l15   = lane & 15;     // M (A-frag) or N (B-frag) within tile
    const int mBase = blockIdx.y * 128;
    const int nBase = blockIdx.x * 64;
    const int wm    = wave * 16;

    // --- per-thread staging descriptors (3 x 16B chunks per K-step) ---
    const int aRow0 = tid >> 2;            // 0..63
    const int aRow1 = aRow0 + 64;          // 64..127
    const int kc    = tid & 3;             // 8-f16 chunk within 32-K slab
    const int bRowL = tid >> 2;            // 0..63 (n within tile)
    int bRowG = nBase + bRowL;             // clamp OOB rows (ragged last N tile)
    if (bRowG > N - 1) bRowG = N - 1;

    const _Float16* aSrc0 = A  + (size_t)(mBase + aRow0) * K + kc * 8;
    const _Float16* aSrc1 = A  + (size_t)(mBase + aRow1) * K + kc * 8;
    const _Float16* bSrc  = Wt + (size_t)bRowG * K + kc * 8;

    const int aDst0 = aRow0 * 64 + ((kc ^ (aRow0 & 3)) << 4);
    const int aDst1 = aRow1 * 64 + ((kc ^ (aRow1 & 3)) << 4);
    const int bDst  = bRowL * 64 + ((kc ^ (bRowL & 3)) << 4);

    auto stage = [&](int kk, int b) {
        async_copy16(aSrc0 + kk, &ldsA[b][aDst0]);
        async_copy16(aSrc1 + kk, &ldsA[b][aDst1]);
        async_copy16(bSrc  + kk, &ldsB[b][bDst]);
    };

    const v8f vzero = {0.f, 0.f, 0.f, 0.f, 0.f, 0.f, 0.f, 0.f};
    v8f acc[4] = {vzero, vzero, vzero, vzero};

    stage(0, 0);
    int buf = 0;
    for (int kk = 0; kk < K; kk += 32) {
        const bool more = (kk + 32) < K;
        if (more) stage(kk + 32, buf ^ 1);     // prefetch next slab (async DMA)
        if (more) { WAIT_ASYNC(3); } else { WAIT_ASYNC(0); }  // current slab done
        __syncthreads();                        // LDS[buf] visible to all waves

        const int rowA = wm + l15;
        h8 alo = *(const h8*)(&ldsA[buf][rowA * 64 + (((g)     ^ (rowA & 3)) << 4)]);
        h8 ahi = *(const h8*)(&ldsA[buf][rowA * 64 + (((2 + g) ^ (rowA & 3)) << 4)]);
        v16h a = __builtin_shufflevector(alo, ahi,
                 0,1,2,3,4,5,6,7,8,9,10,11,12,13,14,15);
#pragma unroll
        for (int s = 0; s < 4; ++s) {
            int nl = s * 16 + l15;
            h8 blo = *(const h8*)(&ldsB[buf][nl * 64 + (((g)     ^ (nl & 3)) << 4)]);
            h8 bhi = *(const h8*)(&ldsB[buf][nl * 64 + (((2 + g) ^ (nl & 3)) << 4)]);
            v16h b = __builtin_shufflevector(blo, bhi,
                     0,1,2,3,4,5,6,7,8,9,10,11,12,13,14,15);
            acc[s] = __builtin_amdgcn_wmma_f32_16x16x32_f16(
                         false, a, false, b, (short)0, acc[s], false, false);
        }
        __syncthreads();                        // all waves done with LDS[buf]
        buf ^= 1;
    }

    // ---- epilogue ----
#pragma unroll
    for (int s = 0; s < 4; ++s) {
        int n_g = nBase + s * 16 + l15;
        if (n_g >= N) continue;
        float bv = bias[n_g];
        if (MODE == 2) {
            int tt  = n_g / FEAT;
            int f   = n_g - tt * FEAT;
            int seg = tt / UPF; if (seg > LOW_T - 2) seg = LOW_T - 2;
            float alpha   = (float)(tt - seg * UPF) * (1.0f / UPF);
            bool anchor   = (tt % UPF) == 0;
            bool interior = tt < (LOW_T - 1) * UPF;
#pragma unroll
            for (int r = 0; r < 8; ++r) {
                int m_g   = mBase + wm + r + 8 * g;
                float dec = acc[s][r] + bv;
                const float* lp = low + (size_t)m_g * (LOW_T * FEAT) + seg * FEAT + f;
                float s0 = lp[0], e0 = lp[FEAT];
                float lin = (1.0f - alpha) * s0 + alpha * e0;
                float bl  = lin + 0.2f * (dec - lin);
                float o   = anchor ? lin : (interior ? bl : dec);
                __builtin_nontemporal_store(o, &Out[(size_t)m_g * ldc + n_g]);
            }
        } else {
#pragma unroll
            for (int r = 0; r < 8; ++r) {
                int m_g = mBase + wm + r + 8 * g;
                float v = acc[s][r] + bv;
                if (MODE == 0) v = fmaxf(v, 0.0f);
                C[(size_t)m_g * ldc + n_g] = (_Float16)v;
            }
        }
    }
}

// ---------------------------------------------------------------------------
extern "C" void kernel_launch(void* const* d_in, const int* in_sizes, int n_in,
                              void* d_out, int out_size, void* d_ws, size_t ws_size,
                              hipStream_t stream)
{
    const float* low    = (const float*)d_in[0];
    const int*   labels = (const int*)  d_in[1];
    const float* W1 = (const float*)d_in[2];  const float* b1 = (const float*)d_in[3];
    const float* W2 = (const float*)d_in[4];  const float* b2 = (const float*)d_in[5];
    const float* W3 = (const float*)d_in[6];  const float* b3 = (const float*)d_in[7];
    const float* W4 = (const float*)d_in[8];  const float* b4 = (const float*)d_in[9];
    const float* W5 = (const float*)d_in[10]; const float* b5 = (const float*)d_in[11];
    const float* W6 = (const float*)d_in[12]; const float* b6 = (const float*)d_in[13];
    const float* emb = (const float*)d_in[14];
    float* out = (float*)d_out;

    char* ws = (char*)d_ws;
    size_t off = 0;
    auto carve = [&](size_t bytes) -> void* {
        void* p = ws + off;
        off += (bytes + 255) & ~(size_t)255;
        return p;
    };

    _Float16* Xh  = (_Float16*)carve((size_t)BATCH * 640 * 2);
    _Float16* W1t = (_Float16*)carve((size_t)512 * 640 * 2);     // N x Kp
    _Float16* W2t = (_Float16*)carve((size_t)256 * 512 * 2);
    _Float16* W3t = (_Float16*)carve((size_t)128 * 256 * 2);
    _Float16* W4t = (_Float16*)carve((size_t)256 * 160 * 2);
    _Float16* W5t = (_Float16*)carve((size_t)512 * 256 * 2);
    _Float16* W6t = (_Float16*)carve((size_t)12000 * 512 * 2);
    _Float16* H1h = (_Float16*)carve((size_t)BATCH * 512 * 2);
    _Float16* H2h = (_Float16*)carve((size_t)BATCH * 256 * 2);
    _Float16* F4h = (_Float16*)carve((size_t)BATCH * 160 * 2);
    _Float16* H4h = (_Float16*)carve((size_t)BATCH * 256 * 2);
    _Float16* H5h = (_Float16*)carve((size_t)BATCH * 512 * 2);

    // activations: X padded 600 -> 640 along K
    {
        int total = BATCH * 640;
        pad_cast_f16<<<(total + 255) / 256, 256, 0, stream>>>(low, Xh, BATCH, 600, BATCH, 640);
    }
    // weights: convert + transpose (+ K padding) -> Wt (N x Kp)
    auto cvtT = [&](const float* s, _Float16* d, int Ks, int N, int Kd) {
        transpose_cast_f16<<<dim3(N / 32, Kd / 32), dim3(32, 8), 0, stream>>>(s, d, Ks, N, Kd);
    };
    cvtT(W1, W1t, 600, 512,   640);
    cvtT(W2, W2t, 512, 256,   512);
    cvtT(W3, W3t, 256, 128,   256);
    cvtT(W4, W4t, 144, 256,   160);
    cvtT(W5, W5t, 256, 512,   256);
    cvtT(W6, W6t, 512, 12000, 512);

    // label embedding into decoder-input columns 128..159
    feat_extra_kernel<<<(BATCH * 32 + 255) / 256, 256, 0, stream>>>(labels, emb, F4h);

    const int mBlocks = BATCH / 128;   // 32
    dim3 blk(256);

    // encoder
    gemm_wmma<0><<<dim3(512 / 64,  mBlocks), blk, 0, stream>>>(Xh,  W1t, b1, H1h, nullptr, nullptr, 512, 640, 512);
    gemm_wmma<0><<<dim3(256 / 64,  mBlocks), blk, 0, stream>>>(H1h, W2t, b2, H2h, nullptr, nullptr, 256, 512, 256);
    gemm_wmma<1><<<dim3(128 / 64,  mBlocks), blk, 0, stream>>>(H2h, W3t, b3, F4h, nullptr, nullptr, 128, 256, 160);
    // decoder
    gemm_wmma<0><<<dim3(256 / 64,  mBlocks), blk, 0, stream>>>(F4h, W4t, b4, H4h, nullptr, nullptr, 256, 160, 256);
    gemm_wmma<0><<<dim3(512 / 64,  mBlocks), blk, 0, stream>>>(H4h, W5t, b5, H5h, nullptr, nullptr, 512, 256, 512);
    // final GEMM with fused interpolation/blend epilogue -> f32 output
    gemm_wmma<2><<<dim3((12000 + 63) / 64, mBlocks), blk, 0, stream>>>(
        H5h, W6t, b6, nullptr, low, out, 12000, 512, 12000);
}
